// FactoredAttention_67293547594149
// MI455X (gfx1250) — matrix operand order repaired and checked by reading, not compiled, not run on hardware
//
#include <hip/hip_runtime.h>

// ---------- types ----------
typedef __bf16        bf16x16 __attribute__((ext_vector_type(16)));
typedef float         f32x8   __attribute__((ext_vector_type(8)));
typedef unsigned int  u32x4   __attribute__((ext_vector_type(4)));
typedef unsigned int  u32x2   __attribute__((ext_vector_type(2)));
typedef int           v4i_g   __attribute__((vector_size(16)));   // matches builtin param
typedef int           v2i_g   __attribute__((vector_size(8)));

union FragU  { bf16x16 v; u32x4 q[2]; };
union Chunk4 { u32x2 q; unsigned short s[4]; };

#define AS1 __attribute__((address_space(1)))
#define AS3 __attribute__((address_space(3)))

#if __has_builtin(__builtin_amdgcn_global_load_async_to_lds_b128) && \
    __has_builtin(__builtin_amdgcn_global_load_async_to_lds_b64)  && \
    __has_builtin(__builtin_amdgcn_s_wait_asynccnt)
#define USE_ASYNC 1
#else
#define USE_ASYNC 0
#endif

#if USE_ASYNC
__device__ __forceinline__ void async_b128(const void* g, void* l) {
  __builtin_amdgcn_global_load_async_to_lds_b128(
      (AS1 v4i_g*)(g), (AS3 v4i_g*)(l), 0, 0);
}
__device__ __forceinline__ void async_b64(const void* g, void* l) {
  __builtin_amdgcn_global_load_async_to_lds_b64(
      (AS1 v2i_g*)(g), (AS3 v2i_g*)(l), 0, 0);
}
#endif

__device__ __forceinline__ unsigned short f2bf(float f) {
  unsigned int u = __float_as_uint(f);
  u += 0x7fffu + ((u >> 16) & 1u);          // round-to-nearest-even
  return (unsigned short)(u >> 16);
}

// ---------- kernel 0a: f32 -> bf16 elementwise ----------
__global__ void cvt_f32_bf16(const float* __restrict__ in,
                             unsigned short* __restrict__ out, int n) {
  int i = blockIdx.x * blockDim.x + threadIdx.x;
  int stride = gridDim.x * blockDim.x;
  for (; i < n; i += stride) out[i] = f2bf(in[i]);
}

// ---------- kernel 0b: f32 KxN -> bf16 NxK (tiled transpose) ----------
__global__ __launch_bounds__(256) void cvt_transpose_bf16(
    const float* __restrict__ in,        // K x N
    unsigned short* __restrict__ out,    // N x K
    int K, int N)
{
  __shared__ unsigned short tile[32][33];
  const int kt = blockIdx.y * 32, nt = blockIdx.x * 32;
  const int c = threadIdx.x & 31, r0 = threadIdx.x >> 5;
  #pragma unroll
  for (int i = 0; i < 4; ++i) {
    int r = r0 + i * 8;
    tile[r][c] = f2bf(in[(size_t)(kt + r) * N + nt + c]);
  }
  __syncthreads();
  #pragma unroll
  for (int i = 0; i < 4; ++i) {
    int r = r0 + i * 8;                  // r -> n, c -> k
    out[(size_t)(nt + r) * K + kt + c] = tile[c][r];
  }
}

// ---------- GEMM: C = A * Bt^T + bias ----------
// A: MxK row-major bf16, Bt: NxK row-major bf16 (pre-transposed weights)
// macro tile 128x128, 256 threads = 8 waves (2M x 4N), wave tile 64x32,
// K-step 32, double-buffered LDS, async global->LDS staging.
template <bool OUT_BF16>
__global__ __launch_bounds__(256) void gemm_bf16(
    const unsigned short* __restrict__ A,
    const unsigned short* __restrict__ Bt,
    const float* __restrict__ bias,
    void* __restrict__ Cout,
    int M, int N, int K)
{
  __shared__ unsigned short Alds[2][128 * 32];  // [m][k]
  __shared__ unsigned short Blds[2][128 * 32];  // [n][k]

  const int tid  = threadIdx.x;
  const int lane = tid & 31;
  const int wave = tid >> 5;
  const int wm   = wave & 1;
  const int wn   = wave >> 1;
  const int lh   = lane >> 4;
  const int ll   = lane & 15;
  const int mblk = blockIdx.y * 128;
  const int nblk = blockIdx.x * 128;

  const f32x8 zero = {0.f,0.f,0.f,0.f,0.f,0.f,0.f,0.f};
  f32x8 acc[4][2];
  #pragma unroll
  for (int i = 0; i < 4; ++i)
    #pragma unroll
    for (int j = 0; j < 2; ++j) acc[i][j] = zero;

  // per-thread stage: 2 x 16B chunks of A + 2 x 16B chunks of B
  auto stage = [&](int buf, int k0) {
    const int c0 = tid * 2;
    #pragma unroll
    for (int i = 0; i < 2; ++i) {
      int c = c0 + i, row = c >> 2, kc = c & 3;
      const unsigned short* ga = A  + (size_t)(mblk + row) * K + k0 + kc * 8;
      const unsigned short* gb = Bt + (size_t)(nblk + row) * K + k0 + kc * 8;
      unsigned short* la = &Alds[buf][row * 32 + kc * 8];
      unsigned short* lb = &Blds[buf][row * 32 + kc * 8];
#if USE_ASYNC
      async_b128(ga, la);
      async_b128(gb, lb);
#else
      *(u32x4*)la = *(const u32x4*)ga;
      *(u32x4*)lb = *(const u32x4*)gb;
#endif
    }
  };

  const int nsteps = K >> 5;
  stage(0, 0);
  for (int s = 0; s < nsteps; ++s) {
    const int cur = s & 1;
    if (s + 1 < nsteps) {
      stage(1 - cur, (s + 1) << 5);
#if USE_ASYNC
      __builtin_amdgcn_s_wait_asynccnt(4);   // current tile's 4 ops landed
#endif
    } else {
#if USE_ASYNC
      __builtin_amdgcn_s_wait_asynccnt(0);
#endif
    }
    __syncthreads();

    FragU a[4], b[2];
    #pragma unroll
    for (int mi = 0; mi < 4; ++mi) {
      const unsigned short* p = &Alds[cur][(wm * 64 + mi * 16 + ll) * 32 + lh * 8];
      a[mi].q[0] = *(const u32x4*)(p);        // K = lh*8 .. +8
      a[mi].q[1] = *(const u32x4*)(p + 16);   // K = 16+lh*8 .. +8
    }
    #pragma unroll
    for (int ni = 0; ni < 2; ++ni) {
      const unsigned short* p = &Blds[cur][(wn * 32 + ni * 16 + ll) * 32 + lh * 8];
      b[ni].q[0] = *(const u32x4*)(p);        // K = lh*8 .. +8
      b[ni].q[1] = *(const u32x4*)(p + 16);
    }
    #pragma unroll
    for (int mi = 0; mi < 4; ++mi)
      #pragma unroll
      for (int ni = 0; ni < 2; ++ni)
        acc[mi][ni] = __builtin_amdgcn_wmma_f32_16x16x32_bf16(
            false, a[mi].v, false, b[ni].v, (short)0, acc[mi][ni], false, false);
    __syncthreads();   // all reads of buf[cur] done before it is re-staged
  }

  // ---- epilogue: bias + store ----
  #pragma unroll
  for (int ni = 0; ni < 2; ++ni) {
    const int col = nblk + wn * 32 + ni * 16 + ll;
    const float bs = bias[col];
    #pragma unroll
    for (int mi = 0; mi < 4; ++mi) {
      #pragma unroll
      for (int r = 0; r < 8; ++r) {
        const int row = mblk + wm * 64 + mi * 16 + lh * 8 + r;
        const float v = acc[mi][ni][r] + bs;
        if (OUT_BF16)
          ((unsigned short*)Cout)[(size_t)row * N + col] = f2bf(v);
        else
          ((float*)Cout)[(size_t)row * N + col] = v;
      }
    }
  }
}

// ---------- attention over block axis ----------
// One WG (4 waves) per (b, h, p): 64x64 scores, causal mask, softmax, W*V.
__global__ __launch_bounds__(128) void attn_kernel(
    const unsigned short* __restrict__ qkv,   // (B, S, 3D) bf16
    unsigned short* __restrict__ obuf)        // (B, S, D)  bf16
{
  const int D = 1024, HD = 64, BL = 128, S = 8192;
  const int idx = blockIdx.x;
  const int b   = idx >> 11;
  const int rem = idx & 2047;
  const int h   = rem >> 7;
  const int p   = rem & 127;

  __shared__ unsigned short Qs [64 * 64];   // [n][d]
  __shared__ unsigned short Ks [64 * 64];   // [m][d]
  __shared__ unsigned short VTs[64 * 64];   // [d][m]
  __shared__ unsigned short Ws [64 * 64];   // [n][m]

  const int tid  = threadIdx.x;
  const int lane = tid & 31;
  const int wave = tid >> 5;
  const int lh   = lane >> 4;
  const int ll   = lane & 15;

  // ---- stage Q, K (async), V^T (scatter) ----
  const size_t baseQ = ((size_t)b * S + p) * (3 * D) + (size_t)h * HD;
  #pragma unroll
  for (int i = 0; i < 8; ++i) {
    int c = tid + i * 128;
    int n = c >> 4;
    int d = (c & 15) * 4;
    size_t g = baseQ + (size_t)n * BL * (3 * D) + d;
#if USE_ASYNC
    async_b64(qkv + g,     Qs + n * 64 + d);
    async_b64(qkv + g + D, Ks + n * 64 + d);
#else
    *(u32x2*)(Qs + n * 64 + d) = *(const u32x2*)(qkv + g);
    *(u32x2*)(Ks + n * 64 + d) = *(const u32x2*)(qkv + g + D);
#endif
    Chunk4 v; v.q = *(const u32x2*)(qkv + g + 2 * D);
    #pragma unroll
    for (int j = 0; j < 4; ++j) VTs[(d + j) * 64 + n] = v.s[j];
  }
#if USE_ASYNC
  __builtin_amdgcn_s_wait_asynccnt(0);
#endif
  __syncthreads();

  const int nbase = wave * 16;
  const f32x8 zero = {0.f,0.f,0.f,0.f,0.f,0.f,0.f,0.f};

  // ---- S = Q K^T ----
  f32x8 sacc[4];
  #pragma unroll
  for (int mt = 0; mt < 4; ++mt) sacc[mt] = zero;
  #pragma unroll
  for (int d0 = 0; d0 < 64; d0 += 32) {
    FragU qa;
    const unsigned short* qp = Qs + (nbase + ll) * 64 + d0 + lh * 8;
    qa.q[0] = *(const u32x4*)(qp);
    qa.q[1] = *(const u32x4*)(qp + 16);
    #pragma unroll
    for (int mt = 0; mt < 4; ++mt) {
      FragU kb;
      const unsigned short* kp = Ks + (mt * 16 + ll) * 64 + d0 + lh * 8;
      kb.q[0] = *(const u32x4*)(kp);
      kb.q[1] = *(const u32x4*)(kp + 16);
      sacc[mt] = __builtin_amdgcn_wmma_f32_16x16x32_bf16(
          false, qa.v, false, kb.v, (short)0, sacc[mt], false, false);
    }
  }

  // ---- scale + causal mask + softmax ----
  #pragma unroll
  for (int r = 0; r < 8; ++r) {
    const int n = nbase + lh * 8 + r;
    float v[4];
    float rmax = -3.4e38f;
    #pragma unroll
    for (int mt = 0; mt < 4; ++mt) {
      const int mcol = mt * 16 + ll;
      float s = sacc[mt][r] * 0.125f;           // 1/sqrt(64)
      if (mcol > n) s = -1.0e9f;
      v[mt] = s;
      rmax = fmaxf(rmax, s);
    }
    #pragma unroll
    for (int off = 1; off < 16; off <<= 1)
      rmax = fmaxf(rmax, __shfl_xor(rmax, off, 32));
    float rsum = 0.f;
    #pragma unroll
    for (int mt = 0; mt < 4; ++mt) { v[mt] = __expf(v[mt] - rmax); rsum += v[mt]; }
    #pragma unroll
    for (int off = 1; off < 16; off <<= 1)
      rsum += __shfl_xor(rsum, off, 32);
    const float inv = 1.0f / rsum;
    #pragma unroll
    for (int mt = 0; mt < 4; ++mt)
      Ws[n * 64 + mt * 16 + ll] = f2bf(v[mt] * inv);
  }
  __syncthreads();

  // ---- O = W V ----
  f32x8 oacc[4];
  #pragma unroll
  for (int dt = 0; dt < 4; ++dt) oacc[dt] = zero;
  #pragma unroll
  for (int m0 = 0; m0 < 64; m0 += 32) {
    FragU wa;
    const unsigned short* wp = Ws + (nbase + ll) * 64 + m0 + lh * 8;
    wa.q[0] = *(const u32x4*)(wp);
    wa.q[1] = *(const u32x4*)(wp + 16);
    #pragma unroll
    for (int dt = 0; dt < 4; ++dt) {
      FragU vb;
      const unsigned short* vp = VTs + (dt * 16 + ll) * 64 + m0 + lh * 8;
      vb.q[0] = *(const u32x4*)(vp);
      vb.q[1] = *(const u32x4*)(vp + 16);
      oacc[dt] = __builtin_amdgcn_wmma_f32_16x16x32_bf16(
          false, wa.v, false, vb.v, (short)0, oacc[dt], false, false);
    }
  }

  // ---- store O: (b, n*BL+p, h*HD+d) ----
  const size_t baseO = ((size_t)b * S + p) * D + (size_t)h * HD;
  #pragma unroll
  for (int dt = 0; dt < 4; ++dt) {
    #pragma unroll
    for (int r = 0; r < 8; ++r) {
      const int n = nbase + lh * 8 + r;
      const int d = dt * 16 + ll;
      obuf[baseO + (size_t)n * BL * D + d] = f2bf(oacc[dt][r]);
    }
  }
}

// ---------- host ----------
extern "C" void kernel_launch(void* const* d_in, const int* in_sizes, int n_in,
                              void* d_out, int out_size, void* d_ws, size_t ws_size,
                              hipStream_t stream) {
  const float* x    = (const float*)d_in[0];
  const float* Wqkv = (const float*)d_in[1];
  const float* bqkv = (const float*)d_in[2];
  const float* Wout = (const float*)d_in[3];
  const float* bout = (const float*)d_in[4];
  float* out = (float*)d_out;

  const int Bt = 2, S = 8192, D = 1024;
  const size_t Mtok = (size_t)Bt * S;                     // 16384

  unsigned short* xb   = (unsigned short*)d_ws;           // 16384 x 1024
  unsigned short* wqb  = xb   + Mtok * D;                 // 3072 x 1024 (transposed)
  unsigned short* wob  = wqb  + (size_t)D * 3 * D;        // 1024 x 1024 (transposed)
  unsigned short* qkvb = wob  + (size_t)D * D;            // 16384 x 3072
  unsigned short* ob   = qkvb + Mtok * 3 * D;             // 16384 x 1024

  cvt_f32_bf16<<<2048, 256, 0, stream>>>(x, xb, (int)(Mtok * D));
  cvt_transpose_bf16<<<dim3(3 * D / 32, D / 32), 256, 0, stream>>>(Wqkv, wqb, D, 3 * D);
  cvt_transpose_bf16<<<dim3(D / 32,     D / 32), 256, 0, stream>>>(Wout, wob, D, D);

  gemm_bf16<true ><<<dim3(3 * D / 128, (int)(Mtok / 128)), 256, 0, stream>>>(
      xb, wqb, bqkv, qkvb, (int)Mtok, 3 * D, D);

  attn_kernel<<<4096, 128, 0, stream>>>(qkvb, ob);

  gemm_bf16<false><<<dim3(D / 128, (int)(Mtok / 128)), 256, 0, stream>>>(
      ob, wob, bout, out, (int)Mtok, D, D);
}
